// MultimodalGenerativeCVAEDecoder_25348896981462
// MI455X (gfx1250) — matrix operand order repaired and checked by reading, not compiled, not run on hardware
//
#include <hip/hip_runtime.h>

typedef __attribute__((ext_vector_type(16))) _Float16 v16h;
typedef __attribute__((ext_vector_type(8)))  float    v8f;

// ---------------- problem dims ----------------
constexpr int NS  = 8;
constexpr int Bsz = 1024;
constexpr int PH  = 12;
constexpr int Nrow = NS * Bsz;          // 8192
constexpr int XS  = 256;
constexpr int ZS  = 32;
constexpr int ZXW = ZS + XS;            // 288  (static GRU-input part)
constexpr int INW = ZXW + 2;            // 290
constexpr int RNN = 512;
constexpr int G3  = 3 * RNN;            // 1536
constexpr float DTC = 0.4f;

// ---------------- output layout (flat fp32, concatenated in return order) ----
constexpr size_t OFF_MU   = (size_t)NS * Bsz * PH;            //  98304
constexpr size_t OFF_LS   = OFF_MU + (size_t)NS*Bsz*PH*2;     // 294912
constexpr size_t OFF_CORR = OFF_LS + (size_t)NS*Bsz*PH*2;     // 491520
constexpr size_t OFF_PRED = OFF_CORR + (size_t)NS*Bsz*PH;     // 589824

// ---------------- workspace layout ----------------
constexpr size_t alu(size_t x) { return (x + 255) & ~(size_t)255; }
constexpr size_t SZ_WHH = (size_t)(G3/16) * (RNN/32) * 512 * 2;   // 96*16 frags
constexpr size_t SZ_WIH = (size_t)(G3/16) * (ZXW/32) * 512 * 2;   // 96*9  frags
constexpr size_t SZ_WH0 = (size_t)(RNN/16) * (ZXW/32) * 512 * 2;  // 32*9  frags
constexpr size_t SZ_WIA = (size_t)G3 * 2 * 4;
constexpr size_t SZ_GI  = (size_t)Nrow * G3 * 2;                  // f16
constexpr size_t SZ_H   = (size_t)Nrow * RNN * 4;                 // f32
constexpr size_t O_WHH = 0;
constexpr size_t O_WIH = alu(O_WHH + SZ_WHH);
constexpr size_t O_WH0 = alu(O_WIH + SZ_WIH);
constexpr size_t O_WIA = alu(O_WH0 + SZ_WH0);
constexpr size_t O_GI  = alu(O_WIA + SZ_WIA);
constexpr size_t O_H   = alu(O_GI  + SZ_GI);
constexpr size_t O_AC  = alu(O_H   + SZ_H);

// =====================================================================
// Pack W (row-major [nOut][rowLen] fp32) into B-fragment blocks (f16):
//   dst[((nt*KC + kc)*32 + lane)*16 + j]  holds B(K = kc*32 + (j&7) + (j>=8?16:0) + (lane>=16?8:0),
//                                                 N = nt*16 + (lane&15))  = W[N][K]
// =====================================================================
__global__ void pack_b(const float* __restrict__ W, _Float16* __restrict__ dst,
                       int nTiles, int KC, int rowLen) {
    int e = blockIdx.x * 256 + threadIdx.x;
    int total = nTiles * KC * 512;
    if (e >= total) return;
    int j    = e & 15;
    int lane = (e >> 4) & 31;
    int frag = e >> 9;
    int kc   = frag % KC;
    int nt   = frag / KC;
    int n = nt * 16 + (lane & 15);
    int k = kc * 32 + (j & 7) + ((j & 8) ? 16 : 0) + ((lane & 16) ? 8 : 0);
    dst[e] = (_Float16)W[(size_t)n * rowLen + k];
}

// ---------------- small init: W_ih action columns + a0 = x_r_t0 @ W_sa^T + b_sa
__global__ void misc_init(const float* __restrict__ x_r_t0,
                          const float* __restrict__ W_sa, const float* __restrict__ b_sa,
                          const float* __restrict__ W_ih,
                          float* __restrict__ wia, float* __restrict__ acur) {
    int e = blockIdx.x * 256 + threadIdx.x;
    if (e < G3 * 2) {
        int c = e >> 1, cc = e & 1;
        wia[e] = W_ih[(size_t)c * INW + ZXW + cc];
    }
    int f = e - G3 * 2;
    if (f >= 0 && f < Nrow * 2) {
        int row = f >> 1, c = f & 1, b = row & (Bsz - 1);
        float s = b_sa[c];
#pragma unroll
        for (int k = 0; k < 8; ++k) s += x_r_t0[b * 8 + k] * W_sa[c * 8 + k];
        acur[f] = s;
    }
}

// =====================================================================
// Prep GEMM (done once): for a 16-row tile compute
//   GI16 = zx @ W_ih[:, :288]^T + b_ih   (1536 cols, f16)   [i = 0..11]
//   H    = zx @ W_h0^T + b_h0            (512 cols h0, f32) [i = 0..3]
// Statically split loops (no divergence); B loads double-buffered.
// =====================================================================
__global__ void prep_gemm(const float* __restrict__ x, const float* __restrict__ z,
                          const _Float16* __restrict__ wih_pack,
                          const _Float16* __restrict__ wh0_pack,
                          const float* __restrict__ b_ih, const float* __restrict__ b_h0,
                          _Float16* __restrict__ GI16, float* __restrict__ H) {
    __shared__ __align__(32) _Float16 afrag[9 * 32 * 16];   // 9 KB
    int tid = threadIdx.x, lane = tid & 31, w = tid >> 5;
    int row0 = blockIdx.x * 16;

    // A-fragment pack of zx (16x288, f16)
    for (int e = tid; e < 9 * 32 * 16; e += 256) {
        int j = e & 15, ln = (e >> 4) & 31, kc = e >> 9;
        int M = ln & 15;
        int K = kc * 32 + (j & 7) + ((j & 8) ? 16 : 0) + ((ln & 16) ? 8 : 0);
        int row = row0 + M;
        float v = (K < ZS) ? z[(size_t)row * ZS + K]
                           : x[(size_t)(row & (Bsz - 1)) * XS + (K - ZS)];
        afrag[e] = (_Float16)v;
    }
    __syncthreads();
    const v16h* aLds = (const v16h*)afrag;

    int colLo = lane & 15;
    int rowHi = (lane & 16) ? 8 : 0;

    // ---- GI columns (W_ih static part) ----
    for (int i = 0; i < 12; ++i) {
        int ct = w * 16 + i * 128;
        const _Float16* bp = wih_pack + (size_t)(ct >> 4) * 9 * 512;
        v8f acc = {};
        v16h av = aLds[lane];
        v16h bv = *(const v16h*)(bp + (size_t)lane * 16);
#pragma unroll
        for (int kc = 0; kc < 9; ++kc) {
            v16h aN = {}, bN = {};
            if (kc < 8) {
                aN = aLds[(kc + 1) * 32 + lane];
                bN = *(const v16h*)(bp + ((size_t)(kc + 1) * 32 + lane) * 16);
            }
            acc = __builtin_amdgcn_wmma_f32_16x16x32_f16(false, av, false, bv,
                                                         (short)0, acc, false, false);
            av = aN; bv = bN;
        }
        int col = ct + colLo;
        float bia = b_ih[col];
#pragma unroll
        for (int r = 0; r < 8; ++r)
            GI16[(size_t)(row0 + r + rowHi) * G3 + col] = (_Float16)(acc[r] + bia);
    }

    // ---- h0 columns (W_h0) ----
    for (int i = 0; i < 4; ++i) {
        int hc0 = w * 16 + i * 128;                 // 0..511
        const _Float16* bp = wh0_pack + (size_t)(hc0 >> 4) * 9 * 512;
        v8f acc = {};
        v16h av = aLds[lane];
        v16h bv = *(const v16h*)(bp + (size_t)lane * 16);
#pragma unroll
        for (int kc = 0; kc < 9; ++kc) {
            v16h aN = {}, bN = {};
            if (kc < 8) {
                aN = aLds[(kc + 1) * 32 + lane];
                bN = *(const v16h*)(bp + ((size_t)(kc + 1) * 32 + lane) * 16);
            }
            acc = __builtin_amdgcn_wmma_f32_16x16x32_f16(false, av, false, bv,
                                                         (short)0, acc, false, false);
            av = aN; bv = bN;
        }
        int hc = hc0 + colLo;
        float bh = b_h0[hc];
#pragma unroll
        for (int r = 0; r < 8; ++r)
            H[(size_t)(row0 + r + rowHi) * RNN + hc] = acc[r] + bh;
    }
}

// =====================================================================
// One GRU step for 16 rows per workgroup (8 waves). Wave w owns matched
// (r,z,n) gate column tiles so gates combine in registers. B (and A)
// fragment loads are software-pipelined one iteration ahead of the WMMAs.
// =====================================================================
__global__ void gru_step(int t,
                         const _Float16* __restrict__ whh_pack,
                         const _Float16* __restrict__ GI16,
                         const float* __restrict__ wia, const float* __restrict__ b_hh,
                         const float* __restrict__ W_mu, const float* __restrict__ b_mu,
                         const float* __restrict__ W_ls, const float* __restrict__ b_ls,
                         const float* __restrict__ W_corr, const float* __restrict__ b_corr,
                         const float* __restrict__ eps_step,
                         float* __restrict__ H, float* __restrict__ acur,
                         float* __restrict__ out) {
    __shared__ __align__(32) _Float16 hfrag[16 * 32 * 16];  // 16 KB A frags of h
    __shared__ float hnewLds[16 * RNN];                      // 32 KB
    __shared__ float aLdsBuf[32];
    __shared__ float headRes[16 * 5];
    int tid = threadIdx.x, lane = tid & 31, w = tid >> 5;
    int row0 = blockIdx.x * 16;

    // pack h (fp32 global) into f16 A fragments
    for (int e = tid; e < 16 * 32 * 16; e += 256) {
        int j = e & 15, ln = (e >> 4) & 31, kc = e >> 9;
        int M = ln & 15;
        int K = kc * 32 + (j & 7) + ((j & 8) ? 16 : 0) + ((ln & 16) ? 8 : 0);
        hfrag[e] = (_Float16)H[(size_t)(row0 + M) * RNN + K];
    }
    if (tid < 32) aLdsBuf[tid] = acur[(size_t)row0 * 2 + tid];
    __syncthreads();
    const v16h* aLds = (const v16h*)hfrag;

    int colLo = lane & 15;
    int rowHi = (lane & 16) ? 8 : 0;

    for (int i = 0; i < 4; ++i) {
        int q0  = w * 16 + i * 128;        // hidden-unit column tile base (0..511)
        int ntR = q0 >> 4;
        const _Float16* pR = whh_pack + (size_t)ntR * 16 * 512;
        const _Float16* pZ = pR + (size_t)32 * 16 * 512;     // +32 col tiles
        const _Float16* pN = pR + (size_t)64 * 16 * 512;     // +64 col tiles
        v8f accR = {}, accZ = {}, accN = {};
        v16h av  = aLds[lane];
        v16h brv = *(const v16h*)(pR + (size_t)lane * 16);
        v16h bzv = *(const v16h*)(pZ + (size_t)lane * 16);
        v16h bnv = *(const v16h*)(pN + (size_t)lane * 16);
#pragma unroll
        for (int kc = 0; kc < 16; ++kc) {
            v16h aN = {}, brN = {}, bzN = {}, bnN = {};
            if (kc < 15) {
                size_t o = ((size_t)(kc + 1) * 32 + lane) * 16;
                aN  = aLds[(kc + 1) * 32 + lane];
                brN = *(const v16h*)(pR + o);
                bzN = *(const v16h*)(pZ + o);
                bnN = *(const v16h*)(pN + o);
            }
            accR = __builtin_amdgcn_wmma_f32_16x16x32_f16(false, av, false, brv, (short)0, accR, false, false);
            accZ = __builtin_amdgcn_wmma_f32_16x16x32_f16(false, av, false, bzv, (short)0, accZ, false, false);
            accN = __builtin_amdgcn_wmma_f32_16x16x32_f16(false, av, false, bnv, (short)0, accN, false, false);
            av = aN; brv = brN; bzv = bzN; bnv = bnN;
        }
        int q = q0 + colLo;
        float bhr = b_hh[q], bhz = b_hh[RNN + q], bhn = b_hh[2 * RNN + q];
        float wr0 = wia[q * 2],             wr1 = wia[q * 2 + 1];
        float wz0 = wia[(RNN + q) * 2],     wz1 = wia[(RNN + q) * 2 + 1];
        float wn0 = wia[(2 * RNN + q) * 2], wn1 = wia[(2 * RNN + q) * 2 + 1];
#pragma unroll
        for (int r = 0; r < 8; ++r) {
            int m   = r + rowHi;
            int row = row0 + m;
            float a0 = aLdsBuf[m * 2], a1 = aLdsBuf[m * 2 + 1];
            const _Float16* gi = GI16 + (size_t)row * G3;
            float gr  = (float)gi[q]           + a0 * wr0 + a1 * wr1 + accR[r] + bhr;
            float gz  = (float)gi[RNN + q]     + a0 * wz0 + a1 * wz1 + accZ[r] + bhz;
            float gin = (float)gi[2 * RNN + q] + a0 * wn0 + a1 * wn1;
            float ghn = accN[r] + bhn;
            float rg = 1.f / (1.f + __expf(-gr));
            float ug = 1.f / (1.f + __expf(-gz));
            float ng = tanhf(gin + rg * ghn);
            float hold = H[(size_t)row * RNN + q];
            float hnew = (1.f - ug) * ng + ug * hold;
            H[(size_t)row * RNN + q] = hnew;
            hnewLds[m * RNN + q] = hnew;
        }
    }
    __syncthreads();

    // output heads: 16 rows x 5 dots (mu0,mu1,ls0,ls1,corr), 10 per wave
    for (int d = 0; d < 10; ++d) {
        int id = w * 10 + d;
        int m = id / 5, hh = id % 5;
        const float* wv = (hh < 2) ? (W_mu + hh * RNN)
                        : (hh < 4) ? (W_ls + (hh - 2) * RNN) : W_corr;
        float bias = (hh < 2) ? b_mu[hh] : (hh < 4) ? b_ls[hh - 2] : b_corr[0];
        float s = 0.f;
        for (int qq = lane; qq < RNN; qq += 32) s += hnewLds[m * RNN + qq] * wv[qq];
#pragma unroll
        for (int o = 16; o > 0; o >>= 1) s += __shfl_xor(s, o, 32);
        if (lane == 0) headRes[m * 5 + hh] = s + bias;
    }
    __syncthreads();

    if (tid < 16) {
        int m = tid, row = row0 + m;
        size_t base = (size_t)row * PH + t;
        float mu0 = headRes[m * 5 + 0], mu1 = headRes[m * 5 + 1];
        float ls0 = headRes[m * 5 + 2], ls1 = headRes[m * 5 + 3];
        float rho = tanhf(headRes[m * 5 + 4]);
        out[base] = 1.0f;                                   // log_pis
        out[OFF_MU + base * 2] = mu0;  out[OFF_MU + base * 2 + 1] = mu1;
        out[OFF_LS + base * 2] = ls0;  out[OFF_LS + base * 2 + 1] = ls1;
        out[OFF_CORR + base] = rho;
        float e0 = eps_step[((size_t)t * Nrow + row) * 2];
        float e1 = eps_step[((size_t)t * Nrow + row) * 2 + 1];
        float s0 = __expf(ls0), s1 = __expf(ls1);
        float q2 = fmaxf(1.f - rho * rho, 1e-5f);
        acur[(size_t)row * 2]     = mu0 + s0 * e0;
        acur[(size_t)row * 2 + 1] = mu1 + s1 * (rho * e0 + sqrtf(q2) * e1);
    }
}

// ---------------- final sample + cumsum integration ----------------
__global__ void preds_kernel(const float* __restrict__ p0,
                             const float* __restrict__ eps_final,
                             float* __restrict__ out) {
    int row = blockIdx.x * 256 + threadIdx.x;
    if (row >= Nrow) return;
    int b = row & (Bsz - 1);
    float px = p0[b * 2], py = p0[b * 2 + 1];
    float cx = 0.f, cy = 0.f;
    for (int t = 0; t < PH; ++t) {
        size_t base = (size_t)row * PH + t;
        float mu0 = out[OFF_MU + base * 2], mu1 = out[OFF_MU + base * 2 + 1];
        float ls0 = out[OFF_LS + base * 2], ls1 = out[OFF_LS + base * 2 + 1];
        float rho = out[OFF_CORR + base];
        float e0 = eps_final[base * 2], e1 = eps_final[base * 2 + 1];
        float s0 = __expf(ls0), s1 = __expf(ls1);
        float q2 = fmaxf(1.f - rho * rho, 1e-5f);
        cx += mu0 + s0 * e0;
        cy += mu1 + s1 * (rho * e0 + sqrtf(q2) * e1);
        out[OFF_PRED + base * 2]     = px + DTC * cx;
        out[OFF_PRED + base * 2 + 1] = py + DTC * cy;
    }
}

// =====================================================================
extern "C" void kernel_launch(void* const* d_in, const int* in_sizes, int n_in,
                              void* d_out, int out_size, void* d_ws, size_t ws_size,
                              hipStream_t stream) {
    const float* x        = (const float*)d_in[0];
    const float* x_r_t0   = (const float*)d_in[1];
    const float* z        = (const float*)d_in[2];
    const float* p0       = (const float*)d_in[3];
    const float* eps_step = (const float*)d_in[4];
    const float* eps_fin  = (const float*)d_in[5];
    const float* W_sa  = (const float*)d_in[6];
    const float* b_sa  = (const float*)d_in[7];
    const float* W_ih  = (const float*)d_in[8];
    const float* b_ih  = (const float*)d_in[9];
    const float* W_hh  = (const float*)d_in[10];
    const float* b_hh  = (const float*)d_in[11];
    const float* W_h0  = (const float*)d_in[12];
    const float* b_h0  = (const float*)d_in[13];
    const float* W_mu  = (const float*)d_in[14];
    const float* b_mu  = (const float*)d_in[15];
    const float* W_ls  = (const float*)d_in[16];
    const float* b_ls  = (const float*)d_in[17];
    const float* W_corr = (const float*)d_in[18];
    const float* b_corr = (const float*)d_in[19];

    char* ws = (char*)d_ws;
    _Float16* whh_pack = (_Float16*)(ws + O_WHH);
    _Float16* wih_pack = (_Float16*)(ws + O_WIH);
    _Float16* wh0_pack = (_Float16*)(ws + O_WH0);
    float*    wia      = (float*)(ws + O_WIA);
    _Float16* GI16     = (_Float16*)(ws + O_GI);
    float*    H        = (float*)(ws + O_H);
    float*    acur     = (float*)(ws + O_AC);
    float*    out      = (float*)d_out;

    // weight repack (f32 -> f16 WMMA B-fragments); tiny, L2-resident afterwards
    pack_b<<<(96 * 16 * 512 + 255) / 256, 256, 0, stream>>>(W_hh, whh_pack, 96, 16, RNN);
    pack_b<<<(96 * 9 * 512 + 255) / 256, 256, 0, stream>>>(W_ih, wih_pack, 96, 9, INW);
    pack_b<<<(32 * 9 * 512 + 255) / 256, 256, 0, stream>>>(W_h0, wh0_pack, 32, 9, ZXW);
    misc_init<<<(G3 * 2 + Nrow * 2 + 255) / 256, 256, 0, stream>>>(
        x_r_t0, W_sa, b_sa, W_ih, wia, acur);

    // one-shot: h0 and the time-invariant input preactivation GI
    prep_gemm<<<Nrow / 16, 256, 0, stream>>>(x, z, wih_pack, wh0_pack, b_ih, b_h0, GI16, H);

    // 12 sequential GRU steps (fused GEMM + gates + heads + sampling)
    for (int t = 0; t < PH; ++t)
        gru_step<<<Nrow / 16, 256, 0, stream>>>(t, whh_pack, GI16, wia, b_hh,
                                                W_mu, b_mu, W_ls, b_ls, W_corr, b_corr,
                                                eps_step, H, acur, out);

    // final GMM sample + dt-cumsum integration
    preds_kernel<<<(Nrow + 255) / 256, 256, 0, stream>>>(p0, eps_fin, out);
}
